// GPT_51977694216493
// MI455X (gfx1250) — matrix operand order repaired and checked by reading, not compiled, not run on hardware
//
#include <hip/hip_runtime.h>

// ---------------- problem constants ----------------
#define LYR 8
#define HDS 16
#define CD  1024
#define BSZ 2
#define TT  1024
#define MM  (BSZ * TT)   // 2048 rows
#define C4  (4 * CD)     // 4096
#define DH  64

// ---------------- WMMA types ----------------
typedef __attribute__((ext_vector_type(16))) __bf16 v16bf;
typedef __attribute__((ext_vector_type(8)))  float  v8f;

union Frag {
    v16bf v;
    unsigned int u[8];
};

__device__ __forceinline__ unsigned short f2bf(float f) {
    unsigned int u = __float_as_uint(f);
    u += 0x7FFFu + ((u >> 16) & 1u);   // round-to-nearest-even
    return (unsigned short)(u >> 16);
}

// ---------------- weight transpose + fp32->bf16 convert ----------------
// W: [K,N] row-major fp32 (y = x @ W), Wt: [N,K] row-major bf16.
// 64x64 LDS tile transpose: fully coalesced 128B reads and writes.
#define TDIM 64
__global__ __launch_bounds__(256) void wconvT_kernel(const float* __restrict__ W,
                                                     unsigned short* __restrict__ Wt,
                                                     int K, int N) {
    __shared__ unsigned short tile[TDIM][TDIM + 2];   // stride 66 u16 = 33 dwords (conflict-free)
    const int tx = threadIdx.x & 63;   // fast dim
    const int ty = threadIdx.x >> 6;   // 0..3
    const int n0 = blockIdx.x * TDIM;
    const int k0 = blockIdx.y * TDIM;

#pragma unroll
    for (int r = 0; r < TDIM; r += 4) {
        tile[r + ty][tx] = f2bf(W[(size_t)(k0 + r + ty) * N + (n0 + tx)]);
    }
    __syncthreads();
#pragma unroll
    for (int r = 0; r < TDIM; r += 4) {
        Wt[(size_t)(n0 + r + ty) * K + (k0 + tx)] = tile[tx][r + ty];
    }
}

// ---------------- layernorm over C=1024, one row per block ----------------
__global__ __launch_bounds__(256) void lnorm_kernel(const float* __restrict__ X,
                                                    const float* __restrict__ w,
                                                    const float* __restrict__ b,
                                                    unsigned short* __restrict__ outb,
                                                    float* __restrict__ outf) {
    __shared__ float s1[256];
    __shared__ float s2[256];
    const int row = blockIdx.x;
    const int tid = threadIdx.x;
    const float* x = X + (size_t)row * CD;

    float v0 = x[tid], v1 = x[tid + 256], v2 = x[tid + 512], v3 = x[tid + 768];
    s1[tid] = v0 + v1 + v2 + v3;
    s2[tid] = v0 * v0 + v1 * v1 + v2 * v2 + v3 * v3;
    __syncthreads();
    for (int off = 128; off > 0; off >>= 1) {
        if (tid < off) {
            s1[tid] += s1[tid + off];
            s2[tid] += s2[tid + off];
        }
        __syncthreads();
    }
    const float mu  = s1[0] * (1.0f / (float)CD);
    const float var = s2[0] * (1.0f / (float)CD) - mu * mu;
    const float inv = rsqrtf(var + 1e-5f);

    float vv[4] = {v0, v1, v2, v3};
#pragma unroll
    for (int i = 0; i < 4; ++i) {
        const int col = tid + i * 256;
        const float y = (vv[i] - mu) * inv * w[col] + b[col];
        if (outb) outb[(size_t)row * CD + col] = f2bf(y);
        else      outf[(size_t)row * CD + col] = y;
    }
}

// ---------------- WMMA GEMM: C[M,N] = A[M,K](bf16) @ Wt[N,K]^T(bf16) + bias ----------------
// One wave per 32x64 output block: 2 M-tiles x 4 N-tiles (8 accumulators),
// 8 WMMAs per 6 fragment loads per k-step for better cache-side intensity.
#define MODE_BF16  0   // store bf16 (qkv)
#define MODE_GELU  1   // exact-erf GELU then store bf16 (mlp1)
#define MODE_RESID 2   // residual add into fp32 (proj, mlp2)

__global__ __launch_bounds__(32) void gemm_wmma_kernel(const unsigned short* __restrict__ A,
                                                       const unsigned short* __restrict__ Wt,
                                                       const float* __restrict__ bias,
                                                       float* __restrict__ Cf,
                                                       unsigned short* __restrict__ Cb,
                                                       int Ndim, int Kdim, int mode) {
    const int lane = threadIdx.x;
    const int l15  = lane & 15;
    const int half = lane >> 4;
    const int m0   = blockIdx.y * 32;
    const int n0   = blockIdx.x * 64;

    const unsigned int* arow[2];
    const unsigned int* brow[4];
#pragma unroll
    for (int i = 0; i < 2; ++i)
        arow[i] = (const unsigned int*)(A + (size_t)(m0 + i * 16 + l15) * Kdim);
#pragma unroll
    for (int j = 0; j < 4; ++j)
        brow[j] = (const unsigned int*)(Wt + (size_t)(n0 + j * 16 + l15) * Kdim);

    v8f acc[2][4] = {};

    const int kd2 = Kdim >> 1;                 // dwords per row
    for (int kk2 = 0; kk2 < kd2; kk2 += 16) {  // 32 bf16 per step
        const int baseA = kk2 + 8 * half;      // A frag: K = 16*half + 0..15
        const int baseB = kk2 + 8 * half;      // B frag: K = 16*half + 0..15
        Frag a[2], bfrag[4];
#pragma unroll
        for (int i = 0; i < 2; ++i) {
#pragma unroll
            for (int j = 0; j < 4; ++j) {
                a[i].u[j]     = arow[i][baseA + j];
                a[i].u[4 + j] = arow[i][baseA + 4 + j];
            }
        }
#pragma unroll
        for (int j = 0; j < 4; ++j) {
#pragma unroll
            for (int jj = 0; jj < 8; ++jj) {
                bfrag[j].u[jj] = brow[j][baseB + jj];
            }
        }
        __builtin_prefetch((const void*)(arow[0] + baseA + 16), 0, 1);
        __builtin_prefetch((const void*)(brow[0] + baseB + 16), 0, 1);
        __builtin_prefetch((const void*)(brow[2] + baseB + 16), 0, 1);

#pragma unroll
        for (int i = 0; i < 2; ++i) {
#pragma unroll
            for (int j = 0; j < 4; ++j) {
                acc[i][j] = __builtin_amdgcn_wmma_f32_16x16x32_bf16(
                    false, a[i].v, false, bfrag[j].v, (short)0, acc[i][j], false, false);
            }
        }
    }

#pragma unroll
    for (int i = 0; i < 2; ++i) {
#pragma unroll
        for (int j = 0; j < 4; ++j) {
#pragma unroll
            for (int r = 0; r < 8; ++r) {
                const int row = m0 + i * 16 + 8 * half + r;   // C/D striping
                const int col = n0 + j * 16 + l15;
                const float v = acc[i][j][r] + bias[col];
                const size_t off = (size_t)row * Ndim + col;
                if (mode == MODE_RESID) {
                    Cf[off] += v;
                } else if (mode == MODE_BF16) {
                    Cb[off] = f2bf(v);
                } else {
                    const float g = 0.5f * v * (1.0f + erff(v * 0.70710678118f));
                    Cb[off] = f2bf(g);
                }
            }
        }
    }
}

// ---------------- flash attention: one wave per (b, h, 16-row query block) ----------------
__global__ __launch_bounds__(32) void attn_kernel(const unsigned short* __restrict__ Q,
                                                  const unsigned short* __restrict__ Kmat,
                                                  const unsigned short* __restrict__ V,
                                                  unsigned short* __restrict__ Y) {
    __shared__ __align__(16) unsigned short sP[16 * 32];

    const int lane = threadIdx.x;
    const int l15  = lane & 15;
    const int half = lane >> 4;
    const int ib = (int)(blockIdx.x & 63);          // query block
    const int hh = (int)((blockIdx.x >> 6) & 15);   // head
    const int bb = (int)(blockIdx.x >> 10);         // batch
    const size_t headoff = (size_t)hh * DH;

    // Q fragments (A operand), rows ib*16 + l15, Dh=64 -> two k-chunks of 32
    const unsigned int* qrow =
        (const unsigned int*)(Q + (size_t)(bb * TT + ib * 16 + l15) * CD + headoff);
    Frag qf0, qf1;
    {
        const int ba = 8 * half;
#pragma unroll
        for (int j = 0; j < 4; ++j) {
            qf0.u[j]     = qrow[ba + j];
            qf0.u[4 + j] = qrow[ba + 4 + j];
            qf1.u[j]     = qrow[16 + ba + j];
            qf1.u[4 + j] = qrow[16 + ba + 4 + j];
        }
    }

    v8f Ot[4] = {};
    float mrow[8], lrow[8];
#pragma unroll
    for (int r = 0; r < 8; ++r) { mrow[r] = -1e30f; lrow[r] = 0.0f; }

    const float scale = 0.125f;   // 1/sqrt(64)
    const int pmax = ib >> 1;     // key-block pairs (32 keys each)

    for (int p = 0; p <= pmax; ++p) {
        const int j0 = 2 * p, j1 = 2 * p + 1;
        const unsigned int* krow0 =
            (const unsigned int*)(Kmat + (size_t)(bb * TT + j0 * 16 + l15) * CD + headoff);
        const unsigned int* krow1 =
            (const unsigned int*)(Kmat + (size_t)(bb * TT + j1 * 16 + l15) * CD + headoff);
        Frag k00, k01, k10, k11;   // (key block, Dh chunk)
        {
            const int bk8 = 8 * half;
#pragma unroll
            for (int j = 0; j < 8; ++j) {
                k00.u[j] = krow0[bk8 + j];
                k01.u[j] = krow0[16 + bk8 + j];
                k10.u[j] = krow1[bk8 + j];
                k11.u[j] = krow1[16 + bk8 + j];
            }
        }
        v8f S0 = {}, S1 = {};
        S0 = __builtin_amdgcn_wmma_f32_16x16x32_bf16(false, qf0.v, false, k00.v, (short)0, S0, false, false);
        S0 = __builtin_amdgcn_wmma_f32_16x16x32_bf16(false, qf1.v, false, k01.v, (short)0, S0, false, false);
        S1 = __builtin_amdgcn_wmma_f32_16x16x32_bf16(false, qf0.v, false, k10.v, (short)0, S1, false, false);
        S1 = __builtin_amdgcn_wmma_f32_16x16x32_bf16(false, qf1.v, false, k11.v, (short)0, S1, false, false);

        float alpha[8];
        const int c0 = j0 * 16 + l15;
        const int c1 = j1 * 16 + l15;
#pragma unroll
        for (int r = 0; r < 8; ++r) {
            const int rowg = ib * 16 + 8 * half + r;
            float s0 = S0[r] * scale;
            float s1 = S1[r] * scale;
            if (c0 > rowg) s0 = -1e30f;           // causal mask
            if (c1 > rowg) s1 = -1e30f;
            float m = fmaxf(s0, s1);
#pragma unroll
            for (int off = 8; off > 0; off >>= 1)
                m = fmaxf(m, __shfl_xor(m, off, 16));   // per-row (half-wave) max
            const float mnew = fmaxf(mrow[r], m);
            const float a  = __expf(mrow[r] - mnew);
            const float p0 = __expf(s0 - mnew);
            const float p1 = __expf(s1 - mnew);
            float ps = p0 + p1;
#pragma unroll
            for (int off = 8; off > 0; off >>= 1)
                ps += __shfl_xor(ps, off, 16);          // per-row sum
            lrow[r] = lrow[r] * a + ps;
            mrow[r] = mnew;
            alpha[r] = a;
            const int lr = 8 * half + r;                 // C/D row for this vgpr
            sP[lr * 32 + l15]      = f2bf(p0);
            sP[lr * 32 + 16 + l15] = f2bf(p1);
        }
        __syncthreads();
        Frag pf;   // P as A-fragment [16 x 32]
        {
            const unsigned int* prow = (const unsigned int*)(sP + l15 * 32);
            const int ba = 8 * half;
#pragma unroll
            for (int j = 0; j < 4; ++j) {
                pf.u[j]     = prow[ba + j];
                pf.u[4 + j] = prow[ba + 4 + j];
            }
        }
        __syncthreads();

        // O[16 x 64] += P[16 x 32] @ V[32 x 64]   (4 N-tiles of 16)
        const unsigned short* vbase = V + (size_t)(bb * TT + p * 32) * CD + headoff + l15;
#pragma unroll
        for (int nt = 0; nt < 4; ++nt) {
            Frag vf;   // B-fragment: column dh = nt*16 + l15, K = 16*half + 0..15
            const unsigned short* vb = vbase + nt * 16;
#pragma unroll
            for (int j = 0; j < 8; ++j) {
                const int k0 = 16 * half + 2 * j;
                const unsigned int lo = vb[(size_t)k0 * CD];
                const unsigned int hi = vb[(size_t)(k0 + 1) * CD];
                vf.u[j] = lo | (hi << 16);
            }
            v8f o = Ot[nt];
#pragma unroll
            for (int r = 0; r < 8; ++r) o[r] *= alpha[r];   // rescale old accumulator
            Ot[nt] = __builtin_amdgcn_wmma_f32_16x16x32_bf16(false, pf.v, false, vf.v, (short)0, o, false, false);
        }
    }

    float rinv[8];
#pragma unroll
    for (int r = 0; r < 8; ++r) rinv[r] = 1.0f / lrow[r];

    unsigned short* yb = Y + (size_t)(bb * TT + ib * 16) * CD + headoff;
#pragma unroll
    for (int nt = 0; nt < 4; ++nt) {
#pragma unroll
        for (int r = 0; r < 8; ++r) {
            const int row = 8 * half + r;
            const int col = nt * 16 + l15;
            yb[(size_t)row * CD + col] = f2bf(Ot[nt][r] * rinv[r]);
        }
    }
}

// ---------------- host orchestration ----------------
extern "C" void kernel_launch(void* const* d_in, const int* in_sizes, int n_in,
                              void* d_out, int out_size, void* d_ws, size_t ws_size,
                              hipStream_t stream) {
    (void)in_sizes; (void)n_in; (void)out_size; (void)ws_size;

    const float* x    = (const float*)d_in[0];
    const float* Wq   = (const float*)d_in[1];
    const float* Wk   = (const float*)d_in[2];
    const float* Wv   = (const float*)d_in[3];
    const float* Wp   = (const float*)d_in[4];
    const float* bq   = (const float*)d_in[5];
    const float* bk   = (const float*)d_in[6];
    const float* bv   = (const float*)d_in[7];
    const float* bp   = (const float*)d_in[8];
    const float* ln1w = (const float*)d_in[9];
    const float* ln1b = (const float*)d_in[10];
    const float* ln2w = (const float*)d_in[11];
    const float* ln2b = (const float*)d_in[12];
    const float* W1   = (const float*)d_in[13];
    const float* b1   = (const float*)d_in[14];
    const float* W2   = (const float*)d_in[15];
    const float* b2   = (const float*)d_in[16];
    const float* lnfw = (const float*)d_in[17];
    const float* lnfb = (const float*)d_in[18];

    char* ws = (char*)d_ws;
    const size_t MiB = 1u << 20;
    float*          h    = (float*)(ws);                         //  8 MiB  fp32 residual stream
    unsigned short* a_bf = (unsigned short*)(ws + 8  * MiB);     //  4 MiB  LN output (bf16)
    unsigned short* q_bf = (unsigned short*)(ws + 12 * MiB);     //  4 MiB
    unsigned short* k_bf = (unsigned short*)(ws + 16 * MiB);     //  4 MiB
    unsigned short* v_bf = (unsigned short*)(ws + 20 * MiB);     //  4 MiB
    unsigned short* y_bf = (unsigned short*)(ws + 24 * MiB);     //  4 MiB
    unsigned short* u_bf = (unsigned short*)(ws + 28 * MiB);     // 16 MiB  MLP hidden (bf16)
    unsigned short* wt   = (unsigned short*)(ws + 44 * MiB);     //  8 MiB  transposed bf16 weight

    hipMemcpyAsync(h, x, (size_t)MM * CD * sizeof(float), hipMemcpyDeviceToDevice, stream);

    const size_t CC  = (size_t)CD * CD;
    const size_t C4C = (size_t)CD * C4;
    const dim3 gC (CD / 64, MM / 32);    // (16, 64)
    const dim3 gC4(C4 / 64, MM / 32);    // (64, 64)
    const dim3 cvCC (CD / TDIM, CD / TDIM);    // transpose grids: (N/64, K/64)
    const dim3 cv1  (C4 / TDIM, CD / TDIM);    // W1: K=CD, N=C4
    const dim3 cv2  (CD / TDIM, C4 / TDIM);    // W2: K=C4, N=CD

    for (int l = 0; l < LYR; ++l) {
        // ---- attention block ----
        lnorm_kernel<<<MM, 256, 0, stream>>>(h, ln1w + l * CD, ln1b + l * CD, a_bf, nullptr);

        wconvT_kernel<<<cvCC, 256, 0, stream>>>(Wq + l * CC, wt, CD, CD);
        gemm_wmma_kernel<<<gC, 32, 0, stream>>>(a_bf, wt, bq + l * CD, nullptr, q_bf, CD, CD, MODE_BF16);

        wconvT_kernel<<<cvCC, 256, 0, stream>>>(Wk + l * CC, wt, CD, CD);
        gemm_wmma_kernel<<<gC, 32, 0, stream>>>(a_bf, wt, bk + l * CD, nullptr, k_bf, CD, CD, MODE_BF16);

        wconvT_kernel<<<cvCC, 256, 0, stream>>>(Wv + l * CC, wt, CD, CD);
        gemm_wmma_kernel<<<gC, 32, 0, stream>>>(a_bf, wt, bv + l * CD, nullptr, v_bf, CD, CD, MODE_BF16);

        attn_kernel<<<BSZ * HDS * (TT / 16), 32, 0, stream>>>(q_bf, k_bf, v_bf, y_bf);

        wconvT_kernel<<<cvCC, 256, 0, stream>>>(Wp + l * CC, wt, CD, CD);
        gemm_wmma_kernel<<<gC, 32, 0, stream>>>(y_bf, wt, bp + l * CD, h, nullptr, CD, CD, MODE_RESID);

        // ---- MLP block ----
        lnorm_kernel<<<MM, 256, 0, stream>>>(h, ln2w + l * CD, ln2b + l * CD, a_bf, nullptr);

        wconvT_kernel<<<cv1, 256, 0, stream>>>(W1 + l * C4C, wt, CD, C4);   // [C,4C] -> [4C,C]
        gemm_wmma_kernel<<<gC4, 32, 0, stream>>>(a_bf, wt, b1 + l * C4, nullptr, u_bf, C4, CD, MODE_GELU);

        wconvT_kernel<<<cv2, 256, 0, stream>>>(W2 + l * C4C, wt, C4, CD);   // [4C,C] -> [C,4C]
        gemm_wmma_kernel<<<gC, 32, 0, stream>>>(u_bf, wt, b2 + l * CD, h, nullptr, CD, C4, MODE_RESID);
    }

    lnorm_kernel<<<MM, 256, 0, stream>>>(h, lnfw, lnfb, nullptr, (float*)d_out);
}
